// MambaBlock_42966852829384
// MI455X (gfx1250) — matrix operand order repaired and checked by reading, not compiled, not run on hardware
//
#include <hip/hip_runtime.h>
#include <math.h>

#define B_     4
#define L_     2048
#define DMODEL 64
#define DSTATE 16
#define DCONV  4
#define DINNER 256
#define DTRANK 4
#define ROWS   (B_*L_)               // 8192
#define DBC_N  (DTRANK + 2*DSTATE)   // 36
#define DBC_P  48                    // padded N for x_proj

typedef __attribute__((ext_vector_type(16))) _Float16 v16h;
typedef __attribute__((ext_vector_type(8)))  _Float16 v8h;
typedef __attribute__((ext_vector_type(8)))  float    v8f;
typedef __attribute__((ext_vector_type(4)))  unsigned int u32x4;
typedef __attribute__((ext_vector_type(8)))  int      i32x8;
typedef __attribute__((ext_vector_type(4)))  int      i32x4;

__device__ __forceinline__ float sigmoidf_(float v){ return 1.f/(1.f+__expf(-v)); }

// ---------------------------------------------------------------------------
// TDM: async-load a [rows x cols] f16 tile (row stride = rowStride elements)
// from global into LDS offset 0 (the kernels below have exactly one __shared__
// array, so its static-LDS offset is 0). Descriptor per CDNA5 ISA 8.3/8.4.
// ---------------------------------------------------------------------------
__device__ __forceinline__ void tdm_load_tile_f16(const _Float16* gsrc,
                                                  int rows, int cols, int rowStride)
{
    unsigned long long ga = (unsigned long long)(size_t)gsrc;
    u32x4 g0;
    g0[0] = 1u;                                            // count=1, user descriptor
    g0[1] = 0u;                                            // lds_addr = 0
    g0[2] = (unsigned)(ga & 0xffffffffu);                  // global_addr[31:0]
    g0[3] = (unsigned)((ga >> 32) & 0x1ffffffu) | (2u<<30);// global_addr[56:32] | type=2
    i32x8 g1;
    g1[0] = (int)(1u << 16);                               // wg_mask=0, data_size=1 (2B)
    g1[1] = (int)((unsigned)(cols & 0xffff) << 16);        // tensor_dim0[15:0] @bit48
    g1[2] = (int)(((unsigned)cols >> 16) |
                  ((unsigned)(ROWS & 0xffff) << 16));      // tensor_dim0 hi | tensor_dim1 lo
    g1[3] = (int)(((unsigned)ROWS >> 16) |
                  ((unsigned)(cols & 0xffff) << 16));      // tensor_dim1 hi | tile_dim0
    g1[4] = (int)((unsigned)(rows & 0xffff));              // tile_dim1 | tile_dim2=0
    g1[5] = (int)(unsigned)rowStride;                      // tensor_dim0_stride[31:0]
    g1[6] = 0;                                             // stride hi | dim1_stride lo
    g1[7] = 0;
    i32x4 z4 = {0,0,0,0};
#if __clang_major__ >= 23
    i32x8 z8 = {0,0,0,0,0,0,0,0};
    __builtin_amdgcn_tensor_load_to_lds(g0, g1, z4, z4, z8, 0);
#else
    __builtin_amdgcn_tensor_load_to_lds(g0, g1, z4, z4, 0);
#endif
}

// ---------------------------------------------------------------------------
// WMMA core: A tile (16 x KTOT f16) in LDS, weights Wh row-major [N, KTOT] f16.
// Per lane the ISA layout is two/one contiguous 16B runs -> b128 loads.
// ---------------------------------------------------------------------------
template<int KTOT, int NT>
__device__ __forceinline__ void mma_core(const _Float16* shA,
                                         const _Float16* __restrict__ Wh,
                                         int n0, int lane, v8f acc[NT])
{
    const int half = lane >> 4;
    const int loc  = lane & 15;
#pragma unroll
    for (int k0 = 0; k0 < KTOT; k0 += 32) {
        v8h a0 = *(const v8h*)(shA + loc*KTOT + k0 + half*8);        // K=half*8+0..7
        v8h a1 = *(const v8h*)(shA + loc*KTOT + k0 + 16 + half*8);   // K=16+half*8+0..7
        v16h a = __builtin_shufflevector(a0, a1, 0,1,2,3,4,5,6,7,8,9,10,11,12,13,14,15);
#pragma unroll
        for (int t = 0; t < NT; ++t) {
            const _Float16* wr = Wh + (size_t)(n0 + t*16 + loc)*KTOT + k0 + half*8;
            v8h b0 = *(const v8h*)wr;          // K = half*8 + 0..7
            v8h b1 = *(const v8h*)(wr + 16);   // K = 16 + half*8 + 0..7
            v16h b = __builtin_shufflevector(b0, b1, 0,1,2,3,4,5,6,7,8,9,10,11,12,13,14,15);
            acc[t] = __builtin_amdgcn_wmma_f32_16x16x32_f16(false, a, false, b,
                                                            (short)0, acc[t], false, false);
        }
    }
}

// ---------------------------------------------------------------------------
// Generic GEMM: one wave per block; TDM stages the 16xKTOT A tile into LDS.
// C[M,N] = A[M,KTOT] * Wh[N,KTOT]^T ; grid = (ROWS/16)*NTG blocks of 32.
// ---------------------------------------------------------------------------
template<int KTOT, int NT, int NTG>
__global__ void k_gemm_tdm(const _Float16* __restrict__ A,
                           const _Float16* __restrict__ Wh,
                           float* __restrict__ C, int ldc)
{
    __shared__ _Float16 shA[16*KTOT];
    const int m0 = (blockIdx.x / NTG) * 16;
    const int n0 = (blockIdx.x % NTG) * NT * 16;
    const int lane = threadIdx.x & 31;

    tdm_load_tile_f16(A + (size_t)m0*KTOT, 16, KTOT, KTOT);
    __builtin_amdgcn_s_wait_tensorcnt(0);

    v8f acc[NT];
#pragma unroll
    for (int t = 0; t < NT; ++t) acc[t] = (v8f){};
    mma_core<KTOT, NT>(shA, Wh, n0, lane, acc);

    const int half = lane>>4, loc = lane&15;
#pragma unroll
    for (int t = 0; t < NT; ++t)
#pragma unroll
        for (int r = 0; r < 8; ++r)
            C[(size_t)(m0 + r + half*8)*ldc + n0 + t*16 + loc] = acc[t][r];
}

// ---------------------------------------------------------------------------
// out_proj GEMM + residual + (un)reverse + concat. NT=4 covers all 64 cols.
// ---------------------------------------------------------------------------
template<int KTOT, int NT>
__global__ void k_gemm_out(const _Float16* __restrict__ Y,   // [ROWS,256] f16
                           const _Float16* __restrict__ Wh,  // [64,256] f16
                           const float* __restrict__ x,      // [ROWS,64] original
                           float* __restrict__ out,          // [ROWS,128]
                           int reverse, int colofs)
{
    __shared__ _Float16 shA[16*KTOT];
    const int m0 = blockIdx.x * 16;
    const int lane = threadIdx.x & 31;

    tdm_load_tile_f16(Y + (size_t)m0*KTOT, 16, KTOT, KTOT);
    __builtin_amdgcn_s_wait_tensorcnt(0);

    v8f acc[NT];
#pragma unroll
    for (int t = 0; t < NT; ++t) acc[t] = (v8f){};
    mma_core<KTOT, NT>(shA, Wh, 0, lane, acc);

    const int half = lane>>4, loc = lane&15;
#pragma unroll
    for (int t = 0; t < NT; ++t)
#pragma unroll
        for (int r = 0; r < 8; ++r) {
            int row = m0 + r + half*8;
            int b = row >> 11, l = row & (L_-1);
            int lf = reverse ? (L_-1-l) : l;
            int col = t*16 + loc;
            size_t orow = (size_t)(b*L_ + lf);
            out[orow*(2*DMODEL) + colofs + col] = acc[t][r] + x[orow*DMODEL + col];
        }
}

// ---------------------------------------------------------------------------
// Weight conversion f32 -> f16, zero-padding rows [srcRows, dstRows)
// ---------------------------------------------------------------------------
__global__ void k_cvt(const float* __restrict__ src, _Float16* __restrict__ dst,
                      int srcRows, int cols, int total)
{
    int i = blockIdx.x*blockDim.x + threadIdx.x;
    if (i >= total) return;
    int r = i / cols;
    dst[i] = (r < srcRows) ? (_Float16)src[i] : (_Float16)0.f;
}

// ---------------------------------------------------------------------------
// RMSNorm (optionally over a sequence-reversed view), output f16
// ---------------------------------------------------------------------------
__global__ void k_rmsnorm(const float* __restrict__ x, const float* __restrict__ nw,
                          _Float16* __restrict__ xn, int reverse)
{
    int row = blockIdx.x*blockDim.x + threadIdx.x;
    if (row >= ROWS) return;
    int b = row >> 11, l = row & (L_-1);
    int src = reverse ? (b*L_ + (L_-1-l)) : row;
    const float* xr = x + (size_t)src*DMODEL;
    float ss = 0.f;
#pragma unroll
    for (int k = 0; k < DMODEL; ++k){ float v = xr[k]; ss += v*v; }
    float sc = rsqrtf(ss*(1.f/DMODEL) + 1e-5f);
    _Float16* o = xn + (size_t)row*DMODEL;
#pragma unroll
    for (int k = 0; k < DMODEL; ++k) o[k] = (_Float16)(xr[k]*sc*nw[k]);
}

// ---------------------------------------------------------------------------
// Depthwise causal conv(4) + bias + SiLU (branchless edge handling)
// ---------------------------------------------------------------------------
__global__ void k_conv(const float* __restrict__ xz,           // [ROWS,512]; xi = cols 0..255
                       const float* __restrict__ cw,           // [256,1,4]
                       const float* __restrict__ cb,           // [256]
                       float* __restrict__ xc32, _Float16* __restrict__ xc16)
{
    int idx = blockIdx.x*blockDim.x + threadIdx.x;
    if (idx >= ROWS*DINNER) return;
    int row = idx >> 8, c = idx & 255;
    int l = row & (L_-1);
    int bstart = row - l;                       // first row of this batch
    float acc = cb[c];
#pragma unroll
    for (int j = 0; j < DCONV; ++j) {
        int rowj = row - (DCONV-1) + j;
        int rclamp = rowj > bstart ? rowj : bstart;           // always-valid address
        float wsel = (rowj >= bstart) ? cw[c*DCONV + j] : 0.f; // select, no branch
        acc += wsel * xz[(size_t)rclamp*(2*DINNER) + c];
    }
    float s = acc * sigmoidf_(acc);
    xc32[(size_t)row*DINNER + c] = s;
    xc16[(size_t)row*DINNER + c] = (_Float16)s;
}

// ---------------------------------------------------------------------------
// Selective scan (sequential over L), fused dt_proj + softplus + D + gate
// ---------------------------------------------------------------------------
__global__ void k_scan(const float* __restrict__ dbc,   // [ROWS,48]: dt(4)|B(16)|C(16)|pad
                       const float* __restrict__ xc32,  // [ROWS,256]
                       const float* __restrict__ xz,    // [ROWS,512] (z at col 256+c)
                       const float* __restrict__ dtw,   // [256,4]
                       const float* __restrict__ dtb,   // [256]
                       const float* __restrict__ Alog,  // [256,16]
                       const float* __restrict__ Dp,    // [256]
                       _Float16* __restrict__ yg)       // [ROWS,256] gated output (f16)
{
    __shared__ float sh[DBC_N];
    int c = threadIdx.x;          // one channel per thread
    int b = blockIdx.x;
    float Ac[DSTATE];
#pragma unroll
    for (int n = 0; n < DSTATE; ++n) Ac[n] = -__expf(Alog[c*DSTATE + n]);
    float w0 = dtw[c*4+0], w1 = dtw[c*4+1], w2 = dtw[c*4+2], w3 = dtw[c*4+3];
    float bb = dtb[c], Dc = Dp[c];
    float h[DSTATE];
#pragma unroll
    for (int n = 0; n < DSTATE; ++n) h[n] = 0.f;

    for (int l = 0; l < L_; ++l) {
        int row = b*L_ + l;
        if (threadIdx.x < DBC_N) sh[threadIdx.x] = dbc[(size_t)row*DBC_P + threadIdx.x];
        __syncthreads();
        float dt = sh[0]*w0 + sh[1]*w1 + sh[2]*w2 + sh[3]*w3 + bb;
        dt = (dt > 20.f) ? dt : __logf(1.f + __expf(dt));    // softplus
        float xcv = xc32[(size_t)row*DINNER + c];
        float y = 0.f;
#pragma unroll
        for (int n = 0; n < DSTATE; ++n) {
            float dA = __expf(dt * Ac[n]);
            h[n] = dA*h[n] + dt * sh[DTRANK+n] * xcv;
            y += h[n] * sh[DTRANK+DSTATE+n];
        }
        y += Dc * xcv;
        float zv = xz[(size_t)row*(2*DINNER) + DINNER + c];
        y *= zv * sigmoidf_(zv);
        yg[(size_t)row*DINNER + c] = (_Float16)y;
        __syncthreads();
    }
}

extern "C" void kernel_launch(void* const* d_in, const int* in_sizes, int n_in,
                              void* d_out, int out_size, void* d_ws, size_t ws_size,
                              hipStream_t stream)
{
    const float* x = (const float*)d_in[0];
    float* out = (float*)d_out;

    char* wp = (char*)d_ws;
    auto alloc = [&](size_t bytes) -> void* {
        void* p = (void*)wp;
        wp += (bytes + 255) & ~(size_t)255;
        return p;
    };

    for (int dir = 0; dir < 2; ++dir) {
        const float* nw   = (const float*)d_in[1 + dir*10 + 0];
        const float* inw  = (const float*)d_in[1 + dir*10 + 1];   // [512,64]
        const float* cw   = (const float*)d_in[1 + dir*10 + 2];   // [256,1,4]
        const float* cb   = (const float*)d_in[1 + dir*10 + 3];   // [256]
        const float* xpw  = (const float*)d_in[1 + dir*10 + 4];   // [36,256]
        const float* dtw  = (const float*)d_in[1 + dir*10 + 5];   // [256,4]
        const float* dtb  = (const float*)d_in[1 + dir*10 + 6];   // [256]
        const float* alog = (const float*)d_in[1 + dir*10 + 7];   // [256,16]
        const float* Dp   = (const float*)d_in[1 + dir*10 + 8];   // [256]
        const float* ow   = (const float*)d_in[1 + dir*10 + 9];   // [64,256]

        _Float16* inw_h = (_Float16*)alloc((size_t)512*64*sizeof(_Float16));
        _Float16* xpw_h = (_Float16*)alloc((size_t)DBC_P*256*sizeof(_Float16));
        _Float16* ow_h  = (_Float16*)alloc((size_t)64*256*sizeof(_Float16));
        _Float16* xn    = (_Float16*)alloc((size_t)ROWS*DMODEL*sizeof(_Float16));
        float*    xz    = (float*)   alloc((size_t)ROWS*2*DINNER*sizeof(float));
        float*    xc32  = (float*)   alloc((size_t)ROWS*DINNER*sizeof(float));
        _Float16* xc16  = (_Float16*)alloc((size_t)ROWS*DINNER*sizeof(_Float16));
        float*    dbc   = (float*)   alloc((size_t)ROWS*DBC_P*sizeof(float));
        _Float16* yg    = (_Float16*)alloc((size_t)ROWS*DINNER*sizeof(_Float16));

        // 0) weights -> f16 (x_proj zero-padded 36 -> 48 rows)
        k_cvt<<<(512*64+255)/256, 256, 0, stream>>>(inw, inw_h, 512, 64, 512*64);
        k_cvt<<<(DBC_P*256+255)/256, 256, 0, stream>>>(xpw, xpw_h, DBC_N, 256, DBC_P*256);
        k_cvt<<<(64*256+255)/256, 256, 0, stream>>>(ow, ow_h, 64, 256, 64*256);

        // 1) RMSNorm (reversed view for backward direction)
        k_rmsnorm<<<ROWS/256, 256, 0, stream>>>(x, nw, xn, dir);
        // 2) in_proj: [8192,64] x [512,64]^T -> [8192,512]; NT=8, 4 n-groups
        k_gemm_tdm<64, 8, 4><<<(ROWS/16)*4, 32, 0, stream>>>(xn, inw_h, xz, 2*DINNER);
        // 3) depthwise conv + SiLU
        k_conv<<<ROWS*DINNER/256, 256, 0, stream>>>(xz, cw, cb, xc32, xc16);
        // 4) x_proj: [8192,256] x [48,256]^T -> [8192,48] (rows 36..47 are zero pad)
        k_gemm_tdm<256, 3, 1><<<ROWS/16, 32, 0, stream>>>(xc16, xpw_h, dbc, DBC_P);
        // 5) fused dt_proj + softplus + selective scan + D skip + SiLU(z) gate
        k_scan<<<B_, DINNER, 0, stream>>>(dbc, xc32, xz, dtw, dtb, alog, Dp, yg);
        // 6) out_proj + residual + un-reverse + concat into output half
        k_gemm_out<256, 4><<<ROWS/16, 32, 0, stream>>>(yg, ow_h, x, out, dir, dir*DMODEL);
    }
}
